// BinConv2d_80805514707619
// MI455X (gfx1250) — 8-bit matrix operand order repaired and checked by reading, compiled, not run on hardware
//
#include <hip/hip_runtime.h>
#include <hip/hip_bf16.h>
#include <stdint.h>


typedef __attribute__((ext_vector_type(8))) int v8i;
typedef __attribute__((ext_vector_type(4))) int v4i;
typedef __attribute__((address_space(1))) v4i gv4i;
typedef __attribute__((address_space(3))) v4i lv4i;

#define C_IN   128
#define C_OUT  128
#define H_IMG  56
#define W_IMG  56
#define N_BATCH 64
#define KDIM   1152
#define KTILES 18
#define HW     3136
#define NPIX   (N_BATCH * HW)
#define RM     2
#define APIX   (RM * 16)
#define ABYTES (APIX * KDIM)
#define MBLKS  (NPIX / APIX)
#define BPACK_BYTES (8 * KTILES * 1024)

#if defined(__has_builtin)
#if __has_builtin(__builtin_amdgcn_global_load_async_to_lds_b128) && \
    __has_builtin(__builtin_amdgcn_s_wait_asynccnt)
#define USE_ASYNC_LDS 1
#endif
#endif

__device__ __forceinline__ signed char sgn8(float v) {
    return v > 0.f ? (signed char)1 : (v < 0.f ? (signed char)-1 : (signed char)0);
}

__global__ void zero_stats_kernel(float* stats) {
    stats[threadIdx.x] = 0.f;
}

__global__ __launch_bounds__(256) void binarize_nhwc_kernel(
    const float* __restrict__ x, signed char* __restrict__ xb)
{
    __shared__ __align__(4) signed char tile[W_IMG * C_IN];
    const int tid = threadIdx.x;
    const int nh  = blockIdx.x;
    const int n   = nh / H_IMG;
    const int h   = nh % H_IMG;
    const size_t base = (size_t)n * C_IN * HW + (size_t)h * W_IMG;

    for (int i = tid; i < W_IMG * C_IN; i += 256) {
        const int c = i / W_IMG, w = i % W_IMG;
        tile[w * C_IN + c] = sgn8(x[base + (size_t)c * HW + w]);
    }
    __syncthreads();
    uint32_t* dst = (uint32_t*)(xb + (size_t)nh * W_IMG * C_IN);
    const uint32_t* src = (const uint32_t*)tile;
    for (int i = tid; i < (W_IMG * C_IN) / 4; i += 256)
        dst[i] = src[i];
}

__global__ void pack_weights_kernel(const float* __restrict__ w,
                                    signed char* __restrict__ B) {
    int idx = blockIdx.x * 256 + threadIdx.x;
    if (idx >= BPACK_BYTES) return;
    int nt   = idx / (KTILES * 1024);
    int r    = idx % (KTILES * 1024);
    int kt   = r / 1024;
    int r2   = r % 1024;
    int lane = r2 / 32;
    int j    = r2 % 32;
    int n     = nt * 16 + (lane & 15);
    int khalf = (lane < 16) ? 0 : 16;
    int kk    = kt * 64 + (j >> 3) * 16 + (lane >> 4) * 8 + (j & 7);
    int tap   = kk >> 7;
    int c     = kk & 127;
    B[idx] = sgn8(w[(size_t)n * KDIM + (size_t)c * 9 + tap]);
}

__global__ __launch_bounds__(256) void binconv_wmma_kernel(
    const signed char* __restrict__ xb,
    const signed char* __restrict__ Bw,
    float* __restrict__ y,
    float* __restrict__ stats)
{
    __shared__ __align__(16) signed char As[ABYTES];
    __shared__ float redS[C_OUT];
    __shared__ float redQ[C_OUT];

    const int tid   = threadIdx.x;
    const int wave  = tid >> 5;
    const int lane  = tid & 31;
    const int pbase = blockIdx.x * APIX;

    if (tid < C_OUT)          redS[tid] = 0.f;
    else if (tid < 2 * C_OUT) redQ[tid - C_OUT] = 0.f;

    for (int slot = tid; slot < ABYTES / 16; slot += 256) {
        const int m     = slot / 72;
        const int r     = slot % 72;
        const int tap   = r >> 3;
        const int chunk = r & 7;
        const int p   = pbase + m;
        const int n   = p / HW;
        const int rem = p % HW;
        const int h   = rem / W_IMG;
        const int w   = rem % W_IMG;
        const int hh  = h - 1 + tap / 3;
        const int wp  = w - 1 + tap % 3;
        signed char* dst = &As[m * KDIM + tap * 128 + chunk * 16];
        if ((unsigned)hh < (unsigned)H_IMG && (unsigned)wp < (unsigned)W_IMG) {
            const signed char* src =
                xb + ((((size_t)n * H_IMG + hh) * W_IMG) + wp) * C_IN + chunk * 16;
#ifdef USE_ASYNC_LDS
            gv4i* gsrc = (gv4i*)(uintptr_t)src;
            v4i*  dgen = (v4i*)dst;
            lv4i* ldst = (lv4i*)dgen;
            __builtin_amdgcn_global_load_async_to_lds_b128(gsrc, ldst, 0, 0);
#else
            *(uint4*)dst = *(const uint4*)src;
#endif
        } else {
            uint4 z = {0u, 0u, 0u, 0u};
            *(uint4*)dst = z;
        }
    }
#ifdef USE_ASYNC_LDS
    __builtin_amdgcn_s_wait_asynccnt(0);
#endif
    __syncthreads();

    const int m    = lane & 15;
    const int half = lane >> 4;
    v8i acc0 = {}, acc1 = {};
    const signed char* bbase = Bw + (size_t)wave * (KTILES * 1024) + lane * 32;

    for (int kt = 0; kt < KTILES; ++kt) {
        const uint32_t* bp = (const uint32_t*)(bbase + kt * 1024);
        v8i b;
        b[0] = (int)bp[0]; b[1] = (int)bp[1]; b[2] = (int)bp[2]; b[3] = (int)bp[3];
        b[4] = (int)bp[4]; b[5] = (int)bp[5]; b[6] = (int)bp[6]; b[7] = (int)bp[7];
        if (kt + 1 < KTILES)
            __builtin_prefetch(bbase + (kt + 1) * 1024, 0, 0);

        const int abase = m * KDIM + kt * 64 + half * 8;
        const uint32_t* ap0 = (const uint32_t*)&As[abase];
        const uint32_t* ap1 = (const uint32_t*)&As[abase + 16 * KDIM];
        v8i a0, a1;
        a0[0] = (int)ap0[0];  a0[1] = (int)ap0[1];
        a0[2] = (int)ap0[4];  a0[3] = (int)ap0[5];
        a0[4] = (int)ap0[8];  a0[5] = (int)ap0[9];
        a0[6] = (int)ap0[12]; a0[7] = (int)ap0[13];
        a1[0] = (int)ap1[0];  a1[1] = (int)ap1[1];
        a1[2] = (int)ap1[4];  a1[3] = (int)ap1[5];
        a1[4] = (int)ap1[8];  a1[5] = (int)ap1[9];
        a1[6] = (int)ap1[12]; a1[7] = (int)ap1[13];

        acc0 = __builtin_amdgcn_wmma_i32_16x16x64_iu8(true, a0, true, b, acc0,
                                                      false, false);
        acc1 = __builtin_amdgcn_wmma_i32_16x16x64_iu8(true, a1, true, b, acc1,
                                                      false, false);
    }

    const int chL = lane & 15;
    const int ch  = wave * 16 + chL;
    float s = 0.f, q = 0.f;

    __syncthreads();
    float* sc = (float*)As;
#pragma unroll
    for (int r = 0; r < 8; ++r) {
        const float v0 = (float)acc0[r];
        const float v1 = (float)acc1[r];
        s += v0 + v1;
        q += v0 * v0 + v1 * v1;
        const int mrow = r + 8 * half;
        sc[wave * 544 +       chL * 17 + mrow] = v0;
        sc[wave * 544 + 272 + chL * 17 + mrow] = v1;
    }
    atomicAdd(&redS[ch], s);
    atomicAdd(&redQ[ch], q);
    __syncthreads();

    for (int j = lane; j < 512; j += 32) {
        const int t   = j >> 8;
        const int idx = j & 255;
        const int c16 = idx >> 4;
        const int m16 = idx & 15;
        const float v = sc[wave * 544 + t * 272 + c16 * 17 + m16];
        const int p   = pbase + t * 16 + m16;
        const int n   = p / HW;
        const int rem = p % HW;
        y[((size_t)n * C_OUT + wave * 16 + c16) * HW + rem] = v;
    }
    if (tid < C_OUT) {
        atomicAdd(&stats[tid], redS[tid]);
        atomicAdd(&stats[C_OUT + tid], redQ[tid]);
    }
}

__global__ void bn_apply_kernel(float* __restrict__ y,
                                const float* __restrict__ stats,
                                const float* __restrict__ gamma,
                                const float* __restrict__ beta,
                                int total)
{
    const float inv = 1.f / (float)NPIX;
    for (int i = blockIdx.x * blockDim.x + threadIdx.x; i < total;
         i += gridDim.x * blockDim.x) {
        const int c = (i / HW) % C_OUT;
        const float mean = stats[c] * inv;
        const float var  = stats[C_OUT + c] * inv - mean * mean;
        const float v = (y[i] - mean) * __frsqrt_rn(var + 1e-5f);
        y[i] = v * gamma[c] + beta[c];
    }
}

extern "C" void kernel_launch(void* const* d_in, const int* in_sizes, int n_in,
                              void* d_out, int out_size, void* d_ws, size_t ws_size,
                              hipStream_t stream) {
    const float* x      = (const float*)d_in[0];
    const float* weight = (const float*)d_in[1];
    const float* gamma  = (const float*)d_in[2];
    const float* beta   = (const float*)d_in[3];
    float* out = (float*)d_out;

    float*       stats = (float*)d_ws;
    signed char* Bpack = (signed char*)d_ws + 1024;
    signed char* xb    = (signed char*)d_ws + 1024 + BPACK_BYTES;

    zero_stats_kernel<<<1, 256, 0, stream>>>(stats);

    binarize_nhwc_kernel<<<N_BATCH * H_IMG, 256, 0, stream>>>(x, xb);

    pack_weights_kernel<<<(BPACK_BYTES + 255) / 256, 256, 0, stream>>>(weight, Bpack);

    binconv_wmma_kernel<<<MBLKS, 256, 0, stream>>>(xb, Bpack, out, stats);

    const int total = N_BATCH * C_OUT * HW;
    bn_apply_kernel<<<8192, 256, 0, stream>>>(out, stats, gamma, beta, total);
}
